// MixerDiffAttention_13993003450482
// MI455X (gfx1250) — matrix operand order repaired and checked by reading, not compiled, not run on hardware
//
#include <hip/hip_runtime.h>
#include <math.h>

#define B_SZ 2
#define T_SZ 2048
#define D_MODEL 2048
#define N_HEADS 16
#define N_KV 8
#define HEAD_DIM 128
#define NQKV 4096            // 2048 (q) + 1024 (k) + 1024 (v) output features
#define BT (B_SZ * T_SZ)     // 4096
#define LAMBDA_INIT 0.2f

typedef __bf16 bf16_t;
typedef bf16_t v16bf __attribute__((ext_vector_type(16)));
typedef float  v8f   __attribute__((ext_vector_type(8)));

union FragU { unsigned int u[8]; v16bf v; };

__device__ __forceinline__ unsigned short f2bf(float f) {
  unsigned int u = __float_as_uint(f);
  return (unsigned short)((u + 0x7FFFu + ((u >> 16) & 1u)) >> 16);  // RNE
}
__device__ __forceinline__ float bf2f(unsigned short h) {
  return __uint_as_float(((unsigned int)h) << 16);
}

// LDS byte offset of a generic pointer into a __shared__ object.
__device__ __forceinline__ unsigned lds_addr(const void* p) {
  return (unsigned)(unsigned long long)(const __attribute__((address_space(3))) void*)p;
}
// CDNA5 async global->LDS copy, 16 bytes per lane (ASYNCcnt-tracked).
__device__ __forceinline__ void async_load_b128(unsigned lds_byte_off, const void* g) {
  asm volatile("global_load_async_to_lds_b128 %0, %1, off"
               :: "v"(lds_byte_off), "v"((unsigned long long)g) : "memory");
}
__device__ __forceinline__ void wait_async0() {
  asm volatile("s_wait_asynccnt 0" ::: "memory");
}

// A-operand (16x32 bf16, M x K): lane m = lane&15; lanes 0-15 hold K pairs
// {0..7,16..23}, lanes 16-31 hold {8..15,24..31}.  One u32 = one K-pair.
__device__ __forceinline__ v16bf load_a_lds(const unsigned short* base, int ld_u16, int lane) {
  FragU f; int m = lane & 15, hf = lane >> 4;
  const unsigned int* row = (const unsigned int*)(base + (size_t)m * ld_u16);
#pragma unroll
  for (int v = 0; v < 8; ++v) f.u[v] = row[(v & 3) + ((v >> 2) << 3) + (hf << 2)];
  return f.v;
}
// B-operand (32x16 bf16, K x N): lane n = lane&15; lanes 0-15 K {0..7,16..23},
// lanes 16-31 K {8..15,24..31}.  Source stored [n][k] row-major (paired K contiguous).
__device__ __forceinline__ v16bf load_b_lds(const unsigned short* base, int ld_u16, int lane) {
  FragU f; int n = lane & 15, hf = lane >> 4;
  const unsigned int* row = (const unsigned int*)(base + (size_t)n * ld_u16);
#pragma unroll
  for (int v = 0; v < 8; ++v) f.u[v] = row[(v & 3) + ((v >> 2) << 3) + (hf << 2)];
  return f.v;
}
__device__ __forceinline__ v8f wmma_bf16(v16bf a, v16bf b, v8f c) {
  return __builtin_amdgcn_wmma_f32_16x16x32_bf16(false, a, false, b, (short)0, c, false, false);
}

// ---------------------------------------------------------------- converters
__global__ __launch_bounds__(256) void cvt_x_kernel(const float* __restrict__ x,
                                                    unsigned short* __restrict__ xb) {
  int i = blockIdx.x * 256 + threadIdx.x;
  xb[i] = f2bf(x[i]);
}

__global__ __launch_bounds__(256) void cvt_w_kernel(const float* __restrict__ wq,
                                                    const float* __restrict__ wk,
                                                    const float* __restrict__ wv,
                                                    unsigned short* __restrict__ wb) {
  int i = blockIdx.x * 256 + threadIdx.x;   // over NQKV * D_MODEL
  int row = i >> 11, c = i & 2047;
  float v;
  if (row < 2048)      v = wq[i];
  else if (row < 3072) v = wk[(size_t)(row - 2048) * 2048 + c];
  else                 v = wv[(size_t)(row - 3072) * 2048 + c];
  wb[i] = f2bf(v);
}

// ---------------------------------------------------------------- QKV GEMM
// out[m][n] = sum_k x[m][k] * W[n][k];  M=BT, N=NQKV, K=D_MODEL (bf16 in, f32 out).
// Double-buffered: async global->LDS copies of the next K-slab overlap WMMAs.
__global__ __launch_bounds__(256) void qkv_gemm_kernel(const unsigned short* __restrict__ xb,
                                                       const unsigned short* __restrict__ wb,
                                                       float* __restrict__ out) {
  __shared__ __attribute__((aligned(16))) unsigned short Xs[2][128 * 32];
  __shared__ __attribute__((aligned(16))) unsigned short Wt[2][128 * 32];
  const int tid = threadIdx.x;
  const int lane = tid & 31, wave = tid >> 5;
  const int hf = lane >> 4, col = lane & 15;
  const int m0 = blockIdx.x * 128;
  const int n0 = blockIdx.y * 128;

  // Each thread stages a contiguous 32B (two b128) of X and of W per slab.
  const int r = tid >> 1;                 // tile row (0..127)
  const int cu = (tid & 1) * 8;           // u32 column base (0 or 8) of 16/row
  const unsigned xs0 = lds_addr(&Xs[0][0]);
  const unsigned wt0 = lds_addr(&Wt[0][0]);
  const unsigned stoff = (unsigned)((r * 32 + cu * 2) * 2);   // byte offset in tile
  const unsigned short* gx = xb + (size_t)(m0 + r) * D_MODEL + cu * 2;
  const unsigned short* gw = wb + (size_t)(n0 + r) * D_MODEL + cu * 2;

  v8f acc[8] = {};

  // Preload slab 0.
  async_load_b128(xs0 + stoff,      gx);
  async_load_b128(xs0 + stoff + 16, gx + 8);
  async_load_b128(wt0 + stoff,      gw);
  async_load_b128(wt0 + stoff + 16, gw + 8);
  wait_async0();
  __syncthreads();

  int buf = 0;
  for (int k0 = 0; k0 < D_MODEL; k0 += 32) {
    if (k0 + 32 < D_MODEL) {  // issue next slab while computing this one
      unsigned dx = xs0 + (unsigned)((buf ^ 1) * 128 * 32 * 2) + stoff;
      unsigned dw = wt0 + (unsigned)((buf ^ 1) * 128 * 32 * 2) + stoff;
      const unsigned short* gx2 = gx + k0 + 32;
      const unsigned short* gw2 = gw + k0 + 32;
      async_load_b128(dx,      gx2);
      async_load_b128(dx + 16, gx2 + 8);
      async_load_b128(dw,      gw2);
      async_load_b128(dw + 16, gw2 + 8);
    }
    v16bf a = load_a_lds(&Xs[buf][wave * 16 * 32], 32, lane);
#pragma unroll
    for (int j = 0; j < 8; ++j) {
      v16bf bfr = load_b_lds(&Wt[buf][j * 16 * 32], 32, lane);
      acc[j] = wmma_bf16(a, bfr, acc[j]);
    }
    wait_async0();
    __syncthreads();
    buf ^= 1;
  }
#pragma unroll
  for (int j = 0; j < 8; ++j)
#pragma unroll
    for (int rr = 0; rr < 8; ++rr) {
      int row = m0 + wave * 16 + rr + (hf << 3);
      int cc = n0 + j * 16 + col;
      out[(size_t)row * NQKV + cc] = acc[j][rr];
    }
}

// ------------------------------------------------- RMSNorm + RoPE + scaling
// One wave per head-row. Rows per (b,t): 16 q, 8 k, 8 v.
__global__ __launch_bounds__(128) void norm_rope_kernel(const float* __restrict__ qkv,
                                                        const float* __restrict__ scaler,
                                                        unsigned short* __restrict__ qn,
                                                        unsigned short* __restrict__ kn,
                                                        unsigned short* __restrict__ vn) {
  int gw = blockIdx.x * 4 + (threadIdx.x >> 5);
  int lane = threadIdx.x & 31;
  int bt = gw >> 5;
  int r = gw & 31;
  if (bt >= BT) return;
  int t = bt & (T_SZ - 1);

  if (r < 24) {
    bool isq = r < 16;
    int h = isq ? r : (r - 16);
    const float* p = qkv + (size_t)bt * NQKV + (isq ? h * HEAD_DIM : 2048 + h * HEAD_DIM);
    float x1a = p[lane], x1b = p[lane + 32], x2a = p[lane + 64], x2b = p[lane + 96];
    float ss = x1a * x1a + x1b * x1b + x2a * x2a + x2b * x2b;
#pragma unroll
    for (int mk = 16; mk >= 1; mk >>= 1) ss += __shfl_xor(ss, mk, 32);
    float inv = rsqrtf(ss * (1.0f / 128.0f) + 1.1920929e-7f);
    x1a *= inv; x1b *= inv; x2a *= inv; x2b *= inv;
    const float c0 = 9.210340371976184f / 64.0f;  // ln(10000)/64
    float fa = (float)t * __expf(-c0 * (float)lane);
    float fb = (float)t * __expf(-c0 * (float)(lane + 32));
    float ca = bf2f(f2bf(__cosf(fa))), sa = bf2f(f2bf(__sinf(fa)));
    float cb = bf2f(f2bf(__cosf(fb))), sb = bf2f(f2bf(__sinf(fb)));
    float o1a =  x1a * ca + x2a * sa;
    float o1b =  x1b * cb + x2b * sb;
    float o2a = -x1a * sa + x2a * ca;
    float o2b = -x1b * sb + x2b * cb;
    float g = 1.0f;
    unsigned short* dst;
    if (isq) {
      g = scaler[h] * logf((float)(t + 1));
      dst = qn + (size_t)bt * 2048 + h * HEAD_DIM;
    } else {
      dst = kn + (size_t)bt * 1024 + h * HEAD_DIM;
    }
    dst[lane]      = f2bf(o1a * g);
    dst[lane + 32] = f2bf(o1b * g);
    dst[lane + 64] = f2bf(o2a * g);
    dst[lane + 96] = f2bf(o2b * g);
  } else {
    int h = r - 24;
    const float* p = qkv + (size_t)bt * NQKV + 3072 + h * HEAD_DIM;
    unsigned short* dst = vn + (size_t)bt * 1024 + h * HEAD_DIM;
#pragma unroll
    for (int j = 0; j < 4; ++j) dst[lane + 32 * j] = f2bf(p[lane + 32 * j]);
  }
}

// ---------------------------------------------------------------- lambda
__global__ void lambda_kernel(const float* lq1, const float* lk1,
                              const float* lq2, const float* lk2, float* lam) {
  int lane = threadIdx.x;
  float s1 = lq1[lane] * lk1[lane] + lq1[lane + 32] * lk1[lane + 32];
  float s2 = lq2[lane] * lk2[lane] + lq2[lane + 32] * lk2[lane + 32];
#pragma unroll
  for (int mk = 16; mk >= 1; mk >>= 1) { s1 += __shfl_xor(s1, mk, 32); s2 += __shfl_xor(s2, mk, 32); }
  if (lane == 0) *lam = __expf(s1) - __expf(s2) + LAMBDA_INIT;
}

// --------------------------------------------- differential flash attention
__device__ __forceinline__ void softmax_update(const float* fa, const float* fb,
                                               float* mrun, float* lrun,
                                               v8f* acc, unsigned short* Pw, int lane) {
  const int hf = lane >> 4, col = lane & 15;
#pragma unroll
  for (int r = 0; r < 8; ++r) {
    float bm = fmaxf(fa[r], fb[r]);
#pragma unroll
    for (int mk = 8; mk >= 1; mk >>= 1) bm = fmaxf(bm, __shfl_xor(bm, mk, 32));
    float nm = fmaxf(mrun[r], bm);
    float fac = (nm == -__builtin_inff()) ? 1.0f : __expf(mrun[r] - nm);
    float pa = (fa[r] == -__builtin_inff()) ? 0.0f : __expf(fa[r] - nm);
    float pb = (fb[r] == -__builtin_inff()) ? 0.0f : __expf(fb[r] - nm);
    float rs = pa + pb;
#pragma unroll
    for (int mk = 8; mk >= 1; mk >>= 1) rs += __shfl_xor(rs, mk, 32);
    lrun[r] = lrun[r] * fac + rs;
    mrun[r] = nm;
#pragma unroll
    for (int nt = 0; nt < 16; ++nt) acc[nt][r] *= fac;
    Pw[(r + (hf << 3)) * 32 + col]      = f2bf(pa);
    Pw[(r + (hf << 3)) * 32 + 16 + col] = f2bf(pb);
  }
}

__global__ __launch_bounds__(256) void diff_attn_kernel(const unsigned short* __restrict__ qn,
                                                        const unsigned short* __restrict__ kn,
                                                        const unsigned short* __restrict__ vn,
                                                        const float* __restrict__ lamp,
                                                        float* __restrict__ out) {
  __shared__ __attribute__((aligned(16))) unsigned short K1s[32 * 128];
  __shared__ __attribute__((aligned(16))) unsigned short K2s[32 * 128];
  __shared__ __attribute__((aligned(16))) unsigned short Vs[256 * 32];  // transposed [e][k]
  __shared__ __attribute__((aligned(16))) unsigned short Ps[8][512];    // per-wave 16x32 P

  const int tid = threadIdx.x;
  const int lane = tid & 31, wave = tid >> 5;
  const int hf = lane >> 4, col = lane & 15;
  const int qt = blockIdx.x, h = blockIdx.y, b = blockIdx.z;
  const int qs = qt * 128;
  const int kvh = h >> 1;                 // GQA: kv head for this attention head
  const int row0 = qs + wave * 16;        // this wave's first query row

  // Persistent Q fragments (q1 = head h, q2 = head 8+h), 4 K-chunks each.
  unsigned int q1f[4][8], q2f[4][8];
  {
    const int m = lane & 15;
    const unsigned short* q1p = qn + ((size_t)(b * T_SZ + row0 + m) * N_HEADS + h) * HEAD_DIM;
    const unsigned short* q2p = qn + ((size_t)(b * T_SZ + row0 + m) * N_HEADS + (8 + h)) * HEAD_DIM;
#pragma unroll
    for (int ch = 0; ch < 4; ++ch)
#pragma unroll
      for (int v = 0; v < 8; ++v) {
        int c = (v & 3) + ((v >> 2) << 3) + (hf << 2);
        q1f[ch][v] = ((const unsigned int*)(q1p + ch * 32))[c];
        q2f[ch][v] = ((const unsigned int*)(q2p + ch * 32))[c];
      }
  }

  v8f acc1[16] = {};
  v8f acc2[16] = {};
  float m1[8], l1[8], m2[8], l2[8];
#pragma unroll
  for (int r = 0; r < 8; ++r) { m1[r] = m2[r] = -__builtin_inff(); l1[r] = l2[r] = 0.0f; }

  const int nblk = qs / 32 + 4;           // causal: keys in [0, qs+128)
  const float sc = 0.08838834764831845f;  // 1/sqrt(128)

  // Async staging constants for K tiles (32B contiguous per thread).
  const int rk = tid >> 3;                // key row 0..31
  const int cuk = (tid & 7) * 8;          // u32 col base of 64/row
  const unsigned k1b = lds_addr(K1s), k2b = lds_addr(K2s);
  const unsigned koff = (unsigned)((rk * 128 + cuk * 2) * 2);

  for (int kb0 = 0; kb0 < nblk; ++kb0) {
    const int kb = kb0 * 32;
    __syncthreads();
    // K1/K2 tiles via CDNA5 async global->LDS (row-major [key][dim]).
    {
      const unsigned short* g1 = kn + ((size_t)(b * T_SZ + kb + rk) * N_KV + kvh) * HEAD_DIM + cuk * 2;
      const unsigned short* g2 = kn + ((size_t)(b * T_SZ + kb + rk) * N_KV + 4 + kvh) * HEAD_DIM + cuk * 2;
      async_load_b128(k1b + koff,      g1);
      async_load_b128(k1b + koff + 16, g1 + 8);
      async_load_b128(k2b + koff,      g2);
      async_load_b128(k2b + koff + 16, g2 + 8);
    }
    // V tile transposed into LDS: Vs[e][k], e in [0,256), k in [0,32).
    {
      int kk = tid >> 3;
      int e0 = (tid & 7) * 32;
      const uint4* vg = (const uint4*)(vn + (size_t)(b * T_SZ + kb + kk) * (N_KV * HEAD_DIM)
                                       + kvh * 2 * HEAD_DIM + e0);
#pragma unroll
      for (int j4 = 0; j4 < 4; ++j4) {
        uint4 w4 = vg[j4];
        unsigned wv_[4] = {w4.x, w4.y, w4.z, w4.w};
#pragma unroll
        for (int q = 0; q < 4; ++q) {
          int e = e0 + j4 * 8 + q * 2;
          Vs[e * 32 + kk]       = (unsigned short)(wv_[q] & 0xffffu);
          Vs[(e + 1) * 32 + kk] = (unsigned short)(wv_[q] >> 16);
        }
      }
    }
    wait_async0();
    __syncthreads();
    if (kb > row0 + 15) continue;   // fully masked for this wave (barriers stay uniform)

    // ---- scores: S1 = Q1*K1^T, S2 = Q2*K2^T (two 16-key sub-tiles each)
    v8f s1a = {}, s1b = {}, s2a = {}, s2b = {};
#pragma unroll
    for (int ch = 0; ch < 4; ++ch) {
      FragU fa1, fa2;
#pragma unroll
      for (int v = 0; v < 8; ++v) { fa1.u[v] = q1f[ch][v]; fa2.u[v] = q2f[ch][v]; }
      v16bf b0_1 = load_b_lds(K1s + ch * 32, 128, lane);
      v16bf b1_1 = load_b_lds(K1s + 16 * 128 + ch * 32, 128, lane);
      v16bf b0_2 = load_b_lds(K2s + ch * 32, 128, lane);
      v16bf b1_2 = load_b_lds(K2s + 16 * 128 + ch * 32, 128, lane);
      s1a = wmma_bf16(fa1.v, b0_1, s1a);
      s1b = wmma_bf16(fa1.v, b1_1, s1b);
      s2a = wmma_bf16(fa2.v, b0_2, s2a);
      s2b = wmma_bf16(fa2.v, b1_2, s2b);
    }

    // ---- scale + causal mask
    float f1a[8], f1b[8], f2a[8], f2b[8];
#pragma unroll
    for (int r = 0; r < 8; ++r) {
      int rowr = row0 + r + (hf << 3);
      int k0i = kb + col, k1i = kb + 16 + col;
      f1a[r] = (k0i <= rowr) ? s1a[r] * sc : -__builtin_inff();
      f1b[r] = (k1i <= rowr) ? s1b[r] * sc : -__builtin_inff();
      f2a[r] = (k0i <= rowr) ? s2a[r] * sc : -__builtin_inff();
      f2b[r] = (k1i <= rowr) ? s2b[r] * sc : -__builtin_inff();
    }

    // ---- stream 1: online softmax + PV
    softmax_update(f1a, f1b, m1, l1, acc1, &Ps[wave][0], lane);
    {
      v16bf pf = load_a_lds(&Ps[wave][0], 32, lane);
#pragma unroll
      for (int nt = 0; nt < 16; ++nt) {
        v16bf bv = load_b_lds(Vs + nt * 16 * 32, 32, lane);
        acc1[nt] = wmma_bf16(pf, bv, acc1[nt]);
      }
    }
    // ---- stream 2
    softmax_update(f2a, f2b, m2, l2, acc2, &Ps[wave][0], lane);
    {
      v16bf pf = load_a_lds(&Ps[wave][0], 32, lane);
#pragma unroll
      for (int nt = 0; nt < 16; ++nt) {
        v16bf bv = load_b_lds(Vs + nt * 16 * 32, 32, lane);
        acc2[nt] = wmma_bf16(pf, bv, acc2[nt]);
      }
    }
  }

  // ---- epilogue: out = y1/l1 - lam * y2/l2
  const float lam = *lamp;
#pragma unroll
  for (int r = 0; r < 8; ++r) {
    int rowr = row0 + r + (hf << 3);
    float i1 = (l1[r] > 0.0f) ? 1.0f / l1[r] : 0.0f;
    float i2 = (l2[r] > 0.0f) ? 1.0f / l2[r] : 0.0f;
    float* op = out + (size_t)(b * T_SZ + rowr) * 2048 + h * 256;
#pragma unroll
    for (int nt = 0; nt < 16; ++nt)
      op[nt * 16 + col] = acc1[nt][r] * i1 - lam * (acc2[nt][r] * i2);
  }
}

// ---------------------------------------------------------------- launcher
extern "C" void kernel_launch(void* const* d_in, const int* in_sizes, int n_in,
                              void* d_out, int out_size, void* d_ws, size_t ws_size,
                              hipStream_t stream) {
  const float* x   = (const float*)d_in[0];
  const float* Wq  = (const float*)d_in[1];
  const float* Wk  = (const float*)d_in[2];
  const float* Wv  = (const float*)d_in[3];
  const float* lq1 = (const float*)d_in[4];
  const float* lk1 = (const float*)d_in[5];
  const float* lq2 = (const float*)d_in[6];
  const float* lk2 = (const float*)d_in[7];
  const float* scl = (const float*)d_in[8];
  float* out = (float*)d_out;
  char* ws = (char*)d_ws;

  size_t off = 0;
  unsigned short* xb = (unsigned short*)(ws + off); off += (size_t)BT * D_MODEL * 2;   // 16 MB
  unsigned short* wb = (unsigned short*)(ws + off); off += (size_t)NQKV * D_MODEL * 2; // 16 MB
  float* qkv = (float*)(ws + off);                  off += (size_t)BT * NQKV * 4;      // 64 MB
  float* lam = (float*)(ws + off);                  off += 256;
  // q/k/v bf16 alias the x/W staging buffers (dead after the GEMM).
  unsigned short* qn = xb;                          // BT*2048 bf16
  unsigned short* kn = wb;                          // BT*1024 bf16
  unsigned short* vn = wb + (size_t)BT * 1024;      // BT*1024 bf16

  cvt_x_kernel<<<(BT * D_MODEL) / 256, 256, 0, stream>>>(x, xb);
  cvt_w_kernel<<<(NQKV * D_MODEL) / 256, 256, 0, stream>>>(Wq, Wk, Wv, wb);
  qkv_gemm_kernel<<<dim3(BT / 128, NQKV / 128), 256, 0, stream>>>(xb, wb, qkv);
  norm_rope_kernel<<<(BT * 32) / 4, 128, 0, stream>>>(qkv, scl, qn, kn, vn);
  lambda_kernel<<<1, 32, 0, stream>>>(lq1, lk1, lq2, lk2, lam);
  diff_attn_kernel<<<dim3(T_SZ / 128, N_KV, B_SZ), 256, 0, stream>>>(qn, kn, vn, lam, out);
}
